// BinRecurrentGCN_10660108829013
// MI455X (gfx1250) — matrix operand order repaired and checked by reading, not compiled, not run on hardware
//
#include <hip/hip_runtime.h>
#include <hip/hip_bf16.h>

#define NN 50000
#define FF 64
#define HIDW 32
#define TT 8
#define EE 800000
#define PP 800000

typedef __attribute__((ext_vector_type(16))) __bf16 v16bf;
typedef __attribute__((ext_vector_type(8)))  float  v8f;

union Frag {
    v16bf v;
    unsigned short u[16];
    uint4 q[2];
};

__device__ __forceinline__ unsigned short f2bf(float f) {
    unsigned u = __float_as_uint(f);
    u += 0x7FFFu + ((u >> 16) & 1u);          // round-to-nearest-even
    return (unsigned short)(u >> 16);
}

__device__ __forceinline__ float sigmoidf_(float x) {
    return 1.0f / (1.0f + __expf(-x));
}

// ---------------------------------------------------------------- zero
__global__ void zero_kernel(float4* p, int n4) {
    int i = blockIdx.x * blockDim.x + threadIdx.x;
    if (i < n4) p[i] = make_float4(0.f, 0.f, 0.f, 0.f);
}

// ---------------------------------------------------------------- degree
__global__ void deg_kernel(const int* __restrict__ row, float* __restrict__ deg) {
    int e = blockIdx.x * blockDim.x + threadIdx.x;
    if (e < EE) atomicAdd(&deg[row[e]], 1.0f);
}

// deg -> dis, in place
__global__ void dis_kernel(float* __restrict__ deg) {
    int i = blockIdx.x * blockDim.x + threadIdx.x;
    if (i < NN) {
        float d = deg[i];
        deg[i] = (d > 0.f) ? rsqrtf(fmaxf(d, 1.0f)) : 0.f;
    }
}

__global__ void norm_kernel(const int* __restrict__ row, const int* __restrict__ col,
                            const float* __restrict__ dis, float* __restrict__ nrm) {
    int e = blockIdx.x * blockDim.x + threadIdx.x;
    if (e < EE) nrm[e] = -dis[row[e]] * dis[col[e]];
}

// ---------------------------------------------------------------- scatter prop: out[col] += norm * in[row]
template<int LD, int SHIFT>   // LD = row width (64 or 32), SHIFT = log2(LD/4)
__global__ void prop_kernel(const int* __restrict__ row, const int* __restrict__ col,
                            const float* __restrict__ nrm,
                            const float* __restrict__ xin, float* __restrict__ xout) {
    int tid = blockIdx.x * blockDim.x + threadIdx.x;
    int e  = tid >> SHIFT;
    int ch = tid & ((1 << SHIFT) - 1);
    if (e >= EE) return;
    int r = row[e], c = col[e];
    float nm = nrm[e];
    float4 v = *(const float4*)(xin + (size_t)r * LD + ch * 4);
    float* d = xout + (size_t)c * LD + ch * 4;
    atomicAdd(d + 0, nm * v.x);
    atomicAdd(d + 1, nm * v.y);
    atomicAdd(d + 2, nm * v.z);
    atomicAdd(d + 3, nm * v.w);
}

// ---------------------------------------------------------------- weight fragment prep
// wfrag: 72 fragments (n 0..7, mm 0..8), each 512 bf16: [lane][i], ISA B layout:
//   element (lane,i): K = (lane>>4)*8 + (i<8 ? i : 8+i), Ncol = n*16 + (lane&15)
// Chebyshev folding: W0' = W0 - W2, W1' = W1, W2' = 2*W2 (feeds raw p2).
__global__ void wprep_kernel(const float* __restrict__ Wx, const float* __restrict__ Wh,
                             const float* __restrict__ bx, const float* __restrict__ bh,
                             const float* __restrict__ bg,
                             unsigned short* __restrict__ wfrag, float* __restrict__ bias128) {
    int tid = blockIdx.x * blockDim.x + threadIdx.x;
    const int total = 72 * 512;
    if (tid < total) {
        int fidx = tid >> 9;
        int r    = tid & 511;
        int lane = r >> 4;
        int i    = r & 15;
        int n  = fidx / 9;
        int mm = fidx % 9;
        int colg = n * 16 + (lane & 15);        // 0..127
        int g = colg >> 5, j = colg & 31;
        int krow = ((lane >> 4) << 3) + (i < 8 ? i : 8 + i);      // 0..31 within K-block
        float val;
        if (mm < 6) {                           // X path, F=64
            int c  = mm >> 1;
            int kk = mm & 1;
            int f  = kk * 32 + krow;
            float w0 = Wx[((g * 3 + 0) * FF + f) * HIDW + j];
            float w1 = Wx[((g * 3 + 1) * FF + f) * HIDW + j];
            float w2 = Wx[((g * 3 + 2) * FF + f) * HIDW + j];
            val = (c == 0) ? (w0 - w2) : (c == 1) ? w1 : (2.0f * w2);
        } else {                                // H path, F=32
            int c = mm - 6;
            int f = krow;
            float w0 = Wh[((g * 3 + 0) * HIDW + f) * HIDW + j];
            float w1 = Wh[((g * 3 + 1) * HIDW + f) * HIDW + j];
            float w2 = Wh[((g * 3 + 2) * HIDW + f) * HIDW + j];
            val = (c == 0) ? (w0 - w2) : (c == 1) ? w1 : (2.0f * w2);
        }
        wfrag[fidx * 512 + lane * 16 + i] = f2bf(val);
    } else if (tid < total + 128) {
        int col = tid - total;
        int g = col >> 5, j = col & 31;
        bias128[col] = bx[g * 32 + j] + bh[g * 32 + j] + bg[g * 32 + j];
    }
}

// ---------------------------------------------------------------- A fragment load (ISA 16-bit A 16x32 layout)
// lane L: M = m0 + (L&15); element i: K = kkbase + (L>>4)*8 + (i<8 ? i : 16+i-8)
__device__ __forceinline__ void load_afrag(Frag& fr, const float* base, int ld,
                                           int m0, int kkbase, int lane) {
    int m  = m0 + (lane & 15);
    int k0 = kkbase + ((lane >> 4) << 3);
    const float* p = base + (size_t)m * ld + k0;
    float4 r0 = *(const float4*)(p);
    float4 r1 = *(const float4*)(p + 4);
    float4 r2 = *(const float4*)(p + 16);
    float4 r3 = *(const float4*)(p + 20);
    fr.u[0]  = f2bf(r0.x); fr.u[1]  = f2bf(r0.y); fr.u[2]  = f2bf(r0.z); fr.u[3]  = f2bf(r0.w);
    fr.u[4]  = f2bf(r1.x); fr.u[5]  = f2bf(r1.y); fr.u[6]  = f2bf(r1.z); fr.u[7]  = f2bf(r1.w);
    fr.u[8]  = f2bf(r2.x); fr.u[9]  = f2bf(r2.y); fr.u[10] = f2bf(r2.z); fr.u[11] = f2bf(r2.w);
    fr.u[12] = f2bf(r3.x); fr.u[13] = f2bf(r3.y); fr.u[14] = f2bf(r3.z); fr.u[15] = f2bf(r3.w);
}

// ---------------------------------------------------------------- fused gate GEMM: G[N,128] = sum_mm A_mm B_mm + bias
__global__ __launch_bounds__(256)
void gemm_gates_kernel(const float* __restrict__ x,
                       const float* __restrict__ p1x, const float* __restrict__ p2x,
                       const float* __restrict__ H,
                       const float* __restrict__ p1h, const float* __restrict__ p2h,
                       const unsigned short* __restrict__ wfrag,
                       const float* __restrict__ bias128,
                       float* __restrict__ G) {
    int wave = threadIdx.x >> 5;
    int lane = threadIdx.x & 31;
    int tile = blockIdx.x * 8 + wave;
    if (tile >= NN / 16) return;          // wave-uniform: EXEC all-ones inside
    int m0 = tile * 16;

    Frag a[9];
    load_afrag(a[0], x,   FF,   m0, 0,  lane);
    load_afrag(a[1], x,   FF,   m0, 32, lane);
    load_afrag(a[2], p1x, FF,   m0, 0,  lane);
    load_afrag(a[3], p1x, FF,   m0, 32, lane);
    load_afrag(a[4], p2x, FF,   m0, 0,  lane);
    load_afrag(a[5], p2x, FF,   m0, 32, lane);
    load_afrag(a[6], H,   HIDW, m0, 0,  lane);
    load_afrag(a[7], p1h, HIDW, m0, 0,  lane);
    load_afrag(a[8], p2h, HIDW, m0, 0,  lane);

    const uint4* wq = (const uint4*)wfrag;
    #pragma unroll
    for (int n = 0; n < 8; ++n) {
        float bv = bias128[n * 16 + (lane & 15)];
        v8f acc = {bv, bv, bv, bv, bv, bv, bv, bv};
        #pragma unroll
        for (int mm = 0; mm < 9; ++mm) {
            Frag b;
            const uint4* bp = wq + ((n * 9 + mm) * 32 + lane) * 2;
            b.q[0] = bp[0];
            b.q[1] = bp[1];
            acc = __builtin_amdgcn_wmma_f32_16x16x32_bf16(
                false, a[mm].v, false, b.v, (short)0, acc, false, false);
        }
        int col   = n * 16 + (lane & 15);
        int rbase = m0 + ((lane >> 4) << 3);
        #pragma unroll
        for (int r = 0; r < 8; ++r)
            G[(size_t)(rbase + r) * 128 + col] = acc[r];
    }
}

// ---------------------------------------------------------------- LSTM gate elementwise (peepholes)
__global__ void gate_kernel(const float* __restrict__ G, const float* __restrict__ wc,
                            float* __restrict__ H, float* __restrict__ C) {
    int tid = blockIdx.x * blockDim.x + threadIdx.x;
    if (tid >= NN * HIDW) return;
    int n = tid >> 5, j = tid & 31;
    const float* g = G + (size_t)n * 128;
    float c  = C[tid];
    float iv = sigmoidf_(g[j]      + wc[j]      * c);
    float fv = sigmoidf_(g[32 + j] + wc[32 + j] * c);
    float cn = fv * c + iv * tanhf(g[64 + j]);
    float ov = sigmoidf_(g[96 + j] + wc[64 + j] * cn);
    H[tid] = ov * tanhf(cn);
    C[tid] = cn;
}

// ---------------------------------------------------------------- head fc1: h16 = relu(relu(H) @ fc1_w + fc1_b)
__global__ void fc1_kernel(const float* __restrict__ H, const float* __restrict__ w,
                           const float* __restrict__ b, float* __restrict__ h16) {
    int tid = blockIdx.x * blockDim.x + threadIdx.x;
    if (tid >= NN * 16) return;
    int n = tid >> 4, k = tid & 15;
    const float* hr = H + (size_t)n * HIDW;
    float s = b[k];
    #pragma unroll
    for (int j = 0; j < 32; ++j) s += fmaxf(hr[j], 0.f) * w[j * 16 + k];
    h16[tid] = fmaxf(s, 0.f);
}

// ---------------------------------------------------------------- head pair MLP
__global__ __launch_bounds__(256)
void pair_kernel(const int* __restrict__ src, const int* __restrict__ dst,
                 const float* __restrict__ h16,
                 const float* __restrict__ fc2w, const float* __restrict__ fc2b,
                 const float* __restrict__ fc3w, const float* __restrict__ fc3b,
                 const float* __restrict__ bw,   const float* __restrict__ bb,
                 float* __restrict__ out) {
    __shared__ float s2w[512], s3w[128], s2b[16], s3b[8], sbw[8], sbb;
    for (int i = threadIdx.x; i < 512; i += 256) s2w[i] = fc2w[i];
    if (threadIdx.x < 128) s3w[threadIdx.x] = fc3w[threadIdx.x];
    if (threadIdx.x < 16)  s2b[threadIdx.x] = fc2b[threadIdx.x];
    if (threadIdx.x < 8) { s3b[threadIdx.x] = fc3b[threadIdx.x]; sbw[threadIdx.x] = bw[threadIdx.x]; }
    if (threadIdx.x == 0)  sbb = bb[0];
    __syncthreads();

    int p = blockIdx.x * 256 + threadIdx.x;
    if (p >= PP) return;
    float u[16], v[16];
    const float4* up = (const float4*)(h16 + (size_t)src[p] * 16);
    const float4* vp = (const float4*)(h16 + (size_t)dst[p] * 16);
    #pragma unroll
    for (int q = 0; q < 4; ++q) {
        float4 a = up[q]; float4 c = vp[q];
        u[q*4+0]=a.x; u[q*4+1]=a.y; u[q*4+2]=a.z; u[q*4+3]=a.w;
        v[q*4+0]=c.x; v[q*4+1]=c.y; v[q*4+2]=c.z; v[q*4+3]=c.w;
    }
    float z1[16];
    #pragma unroll
    for (int k = 0; k < 16; ++k) {
        float s = s2b[k];
        #pragma unroll
        for (int i = 0; i < 16; ++i)
            s += u[i] * s2w[i * 16 + k] + v[i] * s2w[(16 + i) * 16 + k];
        z1[k] = fmaxf(s, 0.f);
    }
    float z2[8];
    #pragma unroll
    for (int m = 0; m < 8; ++m) {
        float s = s3b[m];
        #pragma unroll
        for (int k = 0; k < 16; ++k) s += z1[k] * s3w[k * 8 + m];
        z2[m] = fmaxf(s, 0.f);
    }
    float s = sbb;
    #pragma unroll
    for (int m = 0; m < 8; ++m) s += z2[m] * sbw[m];
    out[p] = sigmoidf_(s);
}

// ================================================================ host launcher
extern "C" void kernel_launch(void* const* d_in, const int* in_sizes, int n_in,
                              void* d_out, int out_size, void* d_ws, size_t ws_size,
                              hipStream_t stream) {
    const float* x    = (const float*)d_in[0];
    const int*   ei   = (const int*)  d_in[1];
    const int*   src  = (const int*)  d_in[2];
    const int*   dst  = (const int*)  d_in[3];
    const float* Wx   = (const float*)d_in[4];
    const float* bx   = (const float*)d_in[5];
    const float* Wh   = (const float*)d_in[6];
    const float* bh   = (const float*)d_in[7];
    const float* bg   = (const float*)d_in[8];
    const float* wc   = (const float*)d_in[9];
    const float* fc1w = (const float*)d_in[10];
    const float* fc1b = (const float*)d_in[11];
    const float* fc2w = (const float*)d_in[12];
    const float* fc2b = (const float*)d_in[13];
    const float* fc3w = (const float*)d_in[14];
    const float* fc3b = (const float*)d_in[15];
    const float* bfw  = (const float*)d_in[16];
    const float* bfb  = (const float*)d_in[17];
    float* out = (float*)d_out;

    // workspace layout (f32 units)
    float* ws = (float*)d_ws;
    const size_t N = NN;
    float* deg  = ws;                       // N            ] zeroed per step
    float* p1x  = ws + N;                   // N*64         ]
    float* p2x  = ws + N * 65;              // N*64         ]
    float* p1h  = ws + N * 129;             // N*32         ]
    float* p2h  = ws + N * 161;             // N*32         ]  -> N*193 total
    float* H    = ws + N * 193;             // N*32  (zero once)
    float* C    = ws + N * 225;             // N*32  (zero once)
    float* G    = ws + N * 257;             // N*128
    float* h16  = ws + N * 385;             // N*16
    float* nrm  = ws + N * 401;             // E
    float* bias128 = ws + N * 401 + EE;     // 128
    unsigned short* wfrag = (unsigned short*)(ws + N * 401 + EE + 128); // 72*512 bf16

    const int B = 256;

    // once per launch: weight fragments + bias fold, zero H|C
    wprep_kernel<<<(72 * 512 + 128 + B - 1) / B, B, 0, stream>>>(Wx, Wh, bx, bh, bg, wfrag, bias128);
    zero_kernel<<<(int)(N * 64 / 4 + B - 1) / B, B, 0, stream>>>((float4*)H, (int)(N * 64 / 4));

    for (int t = 0; t < TT; ++t) {
        const int* row = ei + (size_t)t * 2 * EE;
        const int* col = row + EE;

        // zero deg|p1x|p2x|p1h|p2h in one shot
        zero_kernel<<<(int)(N * 193 / 4 + B - 1) / B, B, 0, stream>>>((float4*)deg, (int)(N * 193 / 4));

        deg_kernel<<<(EE + B - 1) / B, B, 0, stream>>>(row, deg);
        dis_kernel<<<(NN + B - 1) / B, B, 0, stream>>>(deg);
        norm_kernel<<<(EE + B - 1) / B, B, 0, stream>>>(row, col, deg, nrm);

        // X path: p1x = L x ; p2x = L p1x   (T2 folded into weights)
        prop_kernel<64, 4><<<(EE * 16 + B - 1) / B, B, 0, stream>>>(row, col, nrm, x,   p1x);
        prop_kernel<64, 4><<<(EE * 16 + B - 1) / B, B, 0, stream>>>(row, col, nrm, p1x, p2x);
        // H path
        prop_kernel<32, 3><<<(EE * 8 + B - 1) / B, B, 0, stream>>>(row, col, nrm, H,   p1h);
        prop_kernel<32, 3><<<(EE * 8 + B - 1) / B, B, 0, stream>>>(row, col, nrm, p1h, p2h);

        // fused all-gate WMMA GEMM -> G[N,128]
        gemm_gates_kernel<<<(NN / 16 + 7) / 8, 256, 0, stream>>>(x, p1x, p2x, H, p1h, p2h,
                                                                 wfrag, bias128, G);
        gate_kernel<<<(NN * HIDW + B - 1) / B, B, 0, stream>>>(G, wc, H, C);
    }

    fc1_kernel<<<(NN * 16 + B - 1) / B, B, 0, stream>>>(H, fc1w, fc1b, h16);
    pair_kernel<<<(PP + B - 1) / B, B, 0, stream>>>(src, dst, h16, fc2w, fc2b,
                                                    fc3w, fc3b, bfw, bfb, out);
}
